// LinearCrossAttention_54812372631770
// MI455X (gfx1250) — matrix operand order repaired and checked by reading, not compiled, not run on hardware
//
#include <hip/hip_runtime.h>
#include <hip/hip_bf16.h>

// ---------------- problem constants (from setup_inputs) ----------------
#define Bn 4
#define Nn 8192
#define Cc 256      // hidden channels = H*D
#define Hh 8
#define Dd 32
#define TOK 32      // tokens per k1 block
#define LN_EPS 1e-5f

typedef __bf16 bf16_t;
typedef bf16_t v16bf __attribute__((ext_vector_type(16)));
typedef bf16_t v8bf  __attribute__((ext_vector_type(8)));
typedef float  v8f   __attribute__((ext_vector_type(8)));

static __device__ __forceinline__ v8f wmma_bf16(v16bf a, v16bf b, v8f c) {
    // D = A(16x32 bf16) x B(32x16 bf16) + C(16x16 f32)
    return __builtin_amdgcn_wmma_f32_16x16x32_bf16(
        /*neg_a=*/false, a, /*neg_b=*/false, b,
        /*c_mod=*/(short)0, c, /*reuse_a=*/false, /*reuse_b=*/false);
}

static __device__ __forceinline__ v16bf cat8(v8bf lo, v8bf hi) {
    return __builtin_shufflevector(lo, hi, 0,1,2,3,4,5,6,7,8,9,10,11,12,13,14,15);
}

// ---------------- k0: Wk/Wv -> bf16 transposed [col(hd)][c] ----------------
__global__ __launch_bounds__(256) void k0_prep(const float* __restrict__ Wk,
                                               const float* __restrict__ Wv,
                                               bf16_t* __restrict__ WkT,
                                               bf16_t* __restrict__ WvT) {
    const int col = blockIdx.x;   // 0..255 (h*D+d)
    const int c   = threadIdx.x;  // 0..255
    WkT[col * Cc + c] = (bf16_t)Wk[c * Cc + col];
    WvT[col * Cc + c] = (bf16_t)Wv[c * Cc + col];
}

// normalize one C/D accumulator tile from registers and store 8 consecutive
// tokens (16B) into the transposed [b][hd][n] bf16 layout.
static __device__ __forceinline__ void ln_store(v8f acc,
                                                const float* __restrict__ mu,
                                                const float* __restrict__ rs,
                                                float g, float be, int mbase,
                                                bf16_t* __restrict__ dst) {
    v8bf o;
    #pragma unroll
    for (int r = 0; r < 8; ++r)
        o[r] = (bf16_t)((acc[r] - mu[mbase + r]) * rs[mbase + r] * g + be);
    *(v8bf*)dst = o;
}

// ---------------- k1: GEMM (U x Wk, U x Wv) + LayerNorm + transposed store --
// block = 256 threads (8 waves), TOK=32 tokens per block.
// wave w computes output columns [w*32, w*32+32) for both K and V,
// for 2 token (M) tiles -> 8 accumulator tiles per wave.
__global__ __launch_bounds__(256) void k1_gemm_ln(const float* __restrict__ U,
                                                  const bf16_t* __restrict__ WkT,
                                                  const bf16_t* __restrict__ WvT,
                                                  const float* __restrict__ gamma,
                                                  const float* __restrict__ beta,
                                                  bf16_t* __restrict__ klnT,
                                                  bf16_t* __restrict__ vlnT) {
    __shared__ bf16_t aU[TOK][Cc];        // 16 KB: U tile in bf16
    __shared__ float  sum_s[2][TOK];      // cross-wave LN partials (k/v)
    __shared__ float  sq_s[2][TOK];
    __shared__ float  mu_s[2][TOK];
    __shared__ float  rs_s[2][TOK];

    const int tid  = threadIdx.x;
    const int lane = tid & 31;
    const int wave = tid >> 5;

    const int tilesPerB = Nn / TOK;
    const int b    = blockIdx.x / tilesPerB;
    const int tile = blockIdx.x % tilesPerB;
    const int tok0 = tile * TOK;

    // zero LN reduction buffers
    if (tid < 2 * TOK) { sum_s[tid >> 5][tid & 31] = 0.f; sq_s[tid >> 5][tid & 31] = 0.f; }

    // --- phase A: load 32x256 U tile, convert fp32 -> bf16 into LDS ---
    const float* Ub = U + ((size_t)b * Nn + tok0) * Cc;
    for (int i = tid; i < TOK * Cc; i += 256)
        aU[i >> 8][i & 255] = (bf16_t)Ub[i];
    __syncthreads();

    // --- phase B: WMMA GEMM, K = 256 in 8 steps of 32 ---
    v8f ak00 = {}, ak01 = {}, ak10 = {}, ak11 = {};   // k: [mtile][ntile]
    v8f av00 = {}, av01 = {}, av10 = {}, av11 = {};   // v: [mtile][ntile]
    const int rowA0 = lane & 15;          // token row, mtile 0
    const int rowA1 = 16 + (lane & 15);   // token row, mtile 1
    const int hi8   = (lane >> 4) * 8;    // A: lanes 16-31 start at K+8
    const int hi16  = (lane >> 4) * 16;   // unused; B loads also use hi8
    const int colA  = wave * 32 + (lane & 15);

    for (int ks = 0; ks < 8; ++ks) {
        const int c0 = ks * 32;
        // A fragments (16x32 bf16): elems 0..7 = K base..+7, 8..15 = K base+16..+23
        v16bf a0 = cat8(*(const v8bf*)&aU[rowA0][c0 + hi8],
                        *(const v8bf*)&aU[rowA0][c0 + 16 + hi8]);
        v16bf a1 = cat8(*(const v8bf*)&aU[rowA1][c0 + hi8],
                        *(const v8bf*)&aU[rowA1][c0 + 16 + hi8]);
        // B fragments (32x16 bf16): lane = column, elems 0..7 = K base..+7, 8..15 = K base+16..+23
        v16bf bk0 = cat8(*(const v8bf*)&WkT[(size_t)colA        * Cc + c0 + hi8], *(const v8bf*)&WkT[(size_t)colA        * Cc + c0 + 16 + hi8]);
        v16bf bk1 = cat8(*(const v8bf*)&WkT[(size_t)(colA + 16) * Cc + c0 + hi8], *(const v8bf*)&WkT[(size_t)(colA + 16) * Cc + c0 + 16 + hi8]);
        v16bf bv0 = cat8(*(const v8bf*)&WvT[(size_t)colA        * Cc + c0 + hi8], *(const v8bf*)&WvT[(size_t)colA        * Cc + c0 + 16 + hi8]);
        v16bf bv1 = cat8(*(const v8bf*)&WvT[(size_t)(colA + 16) * Cc + c0 + hi8], *(const v8bf*)&WvT[(size_t)(colA + 16) * Cc + c0 + 16 + hi8]);
        ak00 = wmma_bf16(a0, bk0, ak00);
        ak01 = wmma_bf16(a0, bk1, ak01);
        ak10 = wmma_bf16(a1, bk0, ak10);
        ak11 = wmma_bf16(a1, bk1, ak11);
        av00 = wmma_bf16(a0, bv0, av00);
        av01 = wmma_bf16(a0, bv1, av01);
        av10 = wmma_bf16(a1, bv0, av10);
        av11 = wmma_bf16(a1, bv1, av11);
    }

    // --- phase C: LN partial sums from registers ---
    // C/D layout: VGPR r, lanes 0-15 -> M=r, lanes 16-31 -> M=r+8.
    // Each lane owns 2 columns (ntile 0/1); token = r + 8*(lane>>4) + 16*mtile.
    const int half = lane >> 4;
    #pragma unroll
    for (int mt = 0; mt < 2; ++mt) {
        #pragma unroll
        for (int r = 0; r < 8; ++r) {
            float k0v = mt ? ak10[r] : ak00[r];
            float k1v = mt ? ak11[r] : ak01[r];
            float v0v = mt ? av10[r] : av00[r];
            float v1v = mt ? av11[r] : av01[r];
            float ks_ = k0v + k1v, ks2 = k0v * k0v + k1v * k1v;
            float vs_ = v0v + v1v, vs2 = v0v * v0v + v1v * v1v;
            // reduce over the 16-lane half (tokens never cross halves)
            #pragma unroll
            for (int m = 1; m < 16; m <<= 1) {
                ks_ += __shfl_xor(ks_, m, 32);
                ks2 += __shfl_xor(ks2, m, 32);
                vs_ += __shfl_xor(vs_, m, 32);
                vs2 += __shfl_xor(vs2, m, 32);
            }
            if ((lane & 15) == 0) {
                const int m = r + 8 * half + 16 * mt;
                atomicAdd(&sum_s[0][m], ks_);
                atomicAdd(&sq_s[0][m],  ks2);
                atomicAdd(&sum_s[1][m], vs_);
                atomicAdd(&sq_s[1][m],  vs2);
            }
        }
    }
    __syncthreads();

    if (tid < 2 * TOK) {
        const int kv = tid >> 5, m = tid & 31;
        const float mu  = sum_s[kv][m] * (1.0f / 256.0f);
        const float var = sq_s[kv][m]  * (1.0f / 256.0f) - mu * mu;
        mu_s[kv][m] = mu;
        rs_s[kv][m] = rsqrtf(fmaxf(var, 0.f) + LN_EPS);
    }
    __syncthreads();

    // --- phase D: normalize in registers, store transposed bf16 [b][hd][n] ---
    // tile (mt, nt): col = wave*32 + nt*16 + (lane&15); tokens = 8*half+16*mt .. +7
    {
        const int col0 = wave * 32 + (lane & 15);
        const int col1 = col0 + 16;
        const float g0 = gamma[col0], be0 = beta[col0];
        const float g1 = gamma[col1], be1 = beta[col1];
        const int mb0 = 8 * half;        // mtile 0
        const int mb1 = 8 * half + 16;   // mtile 1
        bf16_t* k0p = klnT + ((size_t)b * Cc + col0) * Nn + tok0;
        bf16_t* k1p = klnT + ((size_t)b * Cc + col1) * Nn + tok0;
        bf16_t* v0p = vlnT + ((size_t)b * Cc + col0) * Nn + tok0;
        bf16_t* v1p = vlnT + ((size_t)b * Cc + col1) * Nn + tok0;
        ln_store(ak00, mu_s[0], rs_s[0], g0, be0, mb0, k0p + mb0);
        ln_store(ak01, mu_s[0], rs_s[0], g1, be1, mb0, k1p + mb0);
        ln_store(ak10, mu_s[0], rs_s[0], g0, be0, mb1, k0p + mb1);
        ln_store(ak11, mu_s[0], rs_s[0], g1, be1, mb1, k1p + mb1);
        ln_store(av00, mu_s[1], rs_s[1], g0, be0, mb0, v0p + mb0);
        ln_store(av01, mu_s[1], rs_s[1], g1, be1, mb0, v1p + mb0);
        ln_store(av10, mu_s[1], rs_s[1], g0, be0, mb1, v0p + mb1);
        ln_store(av11, mu_s[1], rs_s[1], g1, be1, mb1, v1p + mb1);
    }
}

// ---------------- k2: kv[b,h,d,e] = sum_n k[n,d] * v[n,e] via WMMA ----------
// one wave per block; grid = B*H*SPLIT, each block contracts N/SPLIT tokens.
#define SPLIT 16
__global__ __launch_bounds__(32) void k2_kv(const bf16_t* __restrict__ klnT,
                                            const bf16_t* __restrict__ vlnT,
                                            float* __restrict__ kvacc) {
    const int lane = threadIdx.x & 31;
    const int b   = blockIdx.x / (Hh * SPLIT);
    const int rem = blockIdx.x % (Hh * SPLIT);
    const int h   = rem / SPLIT;
    const int s   = rem % SPLIT;
    const int tokPer = Nn / SPLIT;   // 512
    const int t0 = s * tokPer;

    v8f a00 = {}, a01 = {}, a10 = {}, a11 = {};  // [dtile][etile]
    const int hi8  = (lane >> 4) * 8;
    const int hi16 = (lane >> 4) * 16;
    // row for A (d) and col for B (e) both use lane&15 within the head
    const size_t base0 = ((size_t)b * Cc + h * Dd + (lane & 15)) * Nn;
    const size_t base1 = base0 + (size_t)16 * Nn;

    for (int ks = 0; ks < tokPer / 32; ++ks) {
        const int k0 = t0 + ks * 32;
        v16bf a0 = cat8(*(const v8bf*)&klnT[base0 + k0 + hi8],
                        *(const v8bf*)&klnT[base0 + k0 + 16 + hi8]);
        v16bf a1 = cat8(*(const v8bf*)&klnT[base1 + k0 + hi8],
                        *(const v8bf*)&klnT[base1 + k0 + 16 + hi8]);
        v16bf b0 = cat8(*(const v8bf*)&vlnT[base0 + k0 + hi8],
                        *(const v8bf*)&vlnT[base0 + k0 + 16 + hi8]);
        v16bf b1 = cat8(*(const v8bf*)&vlnT[base1 + k0 + hi8],
                        *(const v8bf*)&vlnT[base1 + k0 + 16 + hi8]);
        a00 = wmma_bf16(a0, b0, a00);
        a01 = wmma_bf16(a0, b1, a01);
        a10 = wmma_bf16(a1, b0, a10);
        a11 = wmma_bf16(a1, b1, a11);
    }

    float* base = kvacc + ((size_t)b * Hh + h) * Dd * Dd;  // [32][32]
    const int n  = lane & 15;
    const int mo = (lane >> 4) * 8;
    #pragma unroll
    for (int r = 0; r < 8; ++r) {
        atomicAdd(base + (mo + r)      * Dd + n,      a00[r]);
        atomicAdd(base + (mo + r)      * Dd + n + 16, a01[r]);
        atomicAdd(base + (mo + r + 16) * Dd + n,      a10[r]);
        atomicAdd(base + (mo + r + 16) * Dd + n + 16, a11[r]);
    }
}

// ---------------- k3: out[b,n,e*H+h] = X[b,n] * M[b,h,e] / N ----------------
__global__ __launch_bounds__(256) void k3_finalize(const float* __restrict__ X,
                                                   const float* __restrict__ Wq,
                                                   const float* __restrict__ kvacc,
                                                   float* __restrict__ out) {
    __shared__ float Ms[Cc];
    __shared__ float Xs[64];
    const int tid    = threadIdx.x;
    const int chunks = Nn / 64;
    const int b  = blockIdx.x / chunks;
    const int n0 = (blockIdx.x % chunks) * 64;

    // out channel c = e*H + h
    const int c = tid;
    const int h = c & (Hh - 1);
    const int e = c >> 3;
    const float* kvh = kvacc + ((size_t)b * Hh + h) * Dd * Dd;
    float m = 0.f;
    #pragma unroll
    for (int d = 0; d < Dd; ++d) m += Wq[h * Dd + d] * kvh[d * Dd + e];
    Ms[c] = m * (1.0f / (float)Nn);
    if (tid < 64) Xs[tid] = X[(size_t)b * Nn + n0 + tid];
    __syncthreads();

    float* op = out + ((size_t)b * Nn + n0) * Cc + c;
    for (int i = 0; i < 64; ++i) op[(size_t)i * Cc] = Xs[i] * Ms[c];
}

// ---------------- launch ----------------
extern "C" void kernel_launch(void* const* d_in, const int* in_sizes, int n_in,
                              void* d_out, int out_size, void* d_ws, size_t ws_size,
                              hipStream_t stream) {
    const float* U     = (const float*)d_in[0];
    const float* X     = (const float*)d_in[1];
    const float* Wq    = (const float*)d_in[2];
    const float* Wk    = (const float*)d_in[3];
    const float* Wv    = (const float*)d_in[4];
    const float* gamma = (const float*)d_in[5];
    const float* beta  = (const float*)d_in[6];
    float* out = (float*)d_out;

    char* ws = (char*)d_ws;
    bf16_t* WkT  = (bf16_t*)(ws);                                  // 128 KB
    bf16_t* WvT  = (bf16_t*)(ws + (128 << 10));                    // 128 KB
    bf16_t* klnT = (bf16_t*)(ws + (256 << 10));                    // 16 MB
    bf16_t* vlnT = (bf16_t*)(ws + (256 << 10) + (16u << 20));      // 16 MB
    float*  kvacc = (float*)(ws + (256 << 10) + (32u << 20));      // 128 KB

    hipMemsetAsync(kvacc, 0, (size_t)Bn * Hh * Dd * Dd * sizeof(float), stream);
    k0_prep<<<Cc, Cc, 0, stream>>>(Wk, Wv, WkT, WvT);
    k1_gemm_ln<<<Bn * (Nn / TOK), 256, 0, stream>>>(U, WkT, WvT, gamma, beta, klnT, vlnT);
    k2_kv<<<Bn * Hh * SPLIT, 32, 0, stream>>>(klnT, vlnT, kvacc);
    k3_finalize<<<Bn * (Nn / 64), 256, 0, stream>>>(X, Wq, kvacc, out);
}
